// LatentAttention_40089224741105
// MI455X (gfx1250) — matrix operand order repaired and checked by reading, not compiled, not run on hardware
//
#include <hip/hip_runtime.h>
#include <stdint.h>

// ---------------------------------------------------------------------------
// MLA latent-attention block for MI455X (gfx1250, wave32, WMMA).
// Compute-bound (~1.1 TFLOP vs ~400MB traffic @23.3TB/s) -> every matmul on
// v_wmma_f32_16x16x32_bf16 (f32 accumulate), softmax in f32.
// GEMM: 256x128 block tile, 64x64 wave tiles (16 WMMA accumulators/wave),
// double-buffered LDS staging; async global->LDS DMA (ASYNCcnt) when the
// toolchain exposes the gfx1250 builtins, synchronous b128 staging otherwise.
// ---------------------------------------------------------------------------

typedef __attribute__((ext_vector_type(16))) __bf16 v16bf;
typedef __attribute__((ext_vector_type(8)))  float  v8f;

#define SEQ   2048
#define DIM   2048     // D == H*Dh == H*Dl == 2048
#define NBAT  4
#define NHEAD 16
#define DHEAD 128
#define MROWS (NBAT * SEQ)   // 8192

#if defined(__AMDGCN__) && \
    __has_builtin(__builtin_amdgcn_global_load_async_to_lds_b128) && \
    __has_builtin(__builtin_amdgcn_s_wait_asynccnt)
#define USE_ASYNC_LDS 1
// exact pointee type from the builtin's prototype: vector_size(16) int
typedef int b128_t __attribute__((vector_size(16)));
typedef __attribute__((address_space(1))) b128_t gas_b128;
typedef __attribute__((address_space(3))) b128_t las_b128;
#define ASYNC_CP_B128(gp, lp) \
  __builtin_amdgcn_global_load_async_to_lds_b128( \
      (gas_b128*)(gp), (las_b128*)(lp), 0, 0)
#define ASYNC_WAIT_ALL() __builtin_amdgcn_s_wait_asynccnt(0)
#else
#define USE_ASYNC_LDS 0
#define ASYNC_WAIT_ALL() ((void)0)
#endif

union Frag16 { v16bf v; uint4 q[2]; unsigned short s[16]; };
union Pack8  { uint4 q; unsigned short s[8]; };

__device__ __forceinline__ unsigned short f32_to_bf16(float f) {
  union { float f; unsigned u; } x; x.f = f;
  unsigned u = x.u;
  u += 0x7FFFu + ((u >> 16) & 1u);          // round to nearest even
  return (unsigned short)(u >> 16);
}

__device__ __forceinline__ v8f v8f_zero() {
  v8f z = {0.f, 0.f, 0.f, 0.f, 0.f, 0.f, 0.f, 0.f};
  return z;
}

// ---------------------------------------------------------------------------
// f32 -> bf16 conversion (grid-stride)
// ---------------------------------------------------------------------------
__global__ void cvt_f32_bf16(const float* __restrict__ in,
                             unsigned short* __restrict__ out, int n) {
  int i = blockIdx.x * blockDim.x + threadIdx.x;
  int stride = gridDim.x * blockDim.x;
  for (; i < n; i += stride) out[i] = f32_to_bf16(in[i]);
}

// ---------------------------------------------------------------------------
// GEMM: C[M,N] = A[M,K] @ B[N,K]^T  (torch Linear layout), bf16 in, f32 acc.
// Block tile 256x128, K staged 32 at a time through double-buffered LDS.
// 8 waves in a 4x2 grid of 64x64 wave tiles.
// ---------------------------------------------------------------------------
template <bool OUT_BF16>
__global__ __launch_bounds__(256)
void gemm_bf16_wmma(const unsigned short* __restrict__ A,
                    const unsigned short* __restrict__ Bm,
                    void* __restrict__ Cout,
                    int M, int N, int K) {
  constexpr int LDT = 48;  // padded stride: 96B rows keep every b128 16B-aligned
  __shared__ unsigned short lds_a[2][256 * LDT];
  __shared__ unsigned short lds_b[2][128 * LDT];

  const int tid   = threadIdx.x;
  const int lane  = tid & 31;
  const int wave  = tid >> 5;
  const int wm    = (wave >> 1) * 64;   // 0,64,128,192
  const int wn    = (wave & 1) * 64;    // 0,64
  const int m0    = blockIdx.y * 256;
  const int n0    = blockIdx.x * 128;
  const int nlane = lane & 15;
  const int hi    = (lane >= 16);
  const int mbase = hi ? 8 : 0;
  const int aoff  = hi ? 8 : 0;         // A-frag K chunk offset (ISA layout)
  const int boff  = hi ? 16 : 0;        // unused; B-frags also load at aoff

  // staging map: each thread owns two A rows and one B row, a 16-elem chunk
  const int sr = tid >> 1;              // 0..127
  const int sc = (tid & 1) * 16;        // 0 or 16
  const unsigned short* ag0 = &A[(size_t)(m0 + sr) * K + sc];
  const unsigned short* ag1 = &A[(size_t)(m0 + 128 + sr) * K + sc];
  const unsigned short* bg  = &Bm[(size_t)(n0 + sr) * K + sc];

  v8f acc[4][4];
  for (int mt = 0; mt < 4; ++mt)
    for (int nt = 0; nt < 4; ++nt) acc[mt][nt] = v8f_zero();

#if !USE_ASYNC_LDS
  uint4 sreg[6];
#endif

  // phase 1: start the global fetch for stage k0 into buffer `buf`
  auto stage_load = [&](int buf, int k0) {
#if USE_ASYNC_LDS
    unsigned short* la0 = &lds_a[buf][sr * LDT + sc];
    unsigned short* la1 = &lds_a[buf][(128 + sr) * LDT + sc];
    unsigned short* lb  = &lds_b[buf][sr * LDT + sc];
    ASYNC_CP_B128(ag0 + k0,     la0);
    ASYNC_CP_B128(ag0 + k0 + 8, la0 + 8);
    ASYNC_CP_B128(ag1 + k0,     la1);
    ASYNC_CP_B128(ag1 + k0 + 8, la1 + 8);
    ASYNC_CP_B128(bg + k0,      lb);
    ASYNC_CP_B128(bg + k0 + 8,  lb + 8);
    (void)buf;
#else
    sreg[0] = *(const uint4*)&ag0[k0];
    sreg[1] = *(const uint4*)&ag0[k0 + 8];
    sreg[2] = *(const uint4*)&ag1[k0];
    sreg[3] = *(const uint4*)&ag1[k0 + 8];
    sreg[4] = *(const uint4*)&bg[k0];
    sreg[5] = *(const uint4*)&bg[k0 + 8];
    if (k0 + 32 < K) {                    // pull the following slab toward L2
      __builtin_prefetch(&ag0[k0 + 32], 0, 1);
      __builtin_prefetch(&bg[k0 + 32], 0, 1);
    }
#endif
  };

  // phase 2 (sync fallback only): commit staged registers to LDS
  auto stage_commit = [&](int buf) {
#if !USE_ASYNC_LDS
    *(uint4*)&lds_a[buf][sr * LDT + sc]             = sreg[0];
    *(uint4*)&lds_a[buf][sr * LDT + sc + 8]         = sreg[1];
    *(uint4*)&lds_a[buf][(128 + sr) * LDT + sc]     = sreg[2];
    *(uint4*)&lds_a[buf][(128 + sr) * LDT + sc + 8] = sreg[3];
    *(uint4*)&lds_b[buf][sr * LDT + sc]             = sreg[4];
    *(uint4*)&lds_b[buf][sr * LDT + sc + 8]         = sreg[5];
#else
    (void)buf;
#endif
  };

  // prologue: fill buffer 0 with stage 0
  stage_load(0, 0);
  stage_commit(0);
  ASYNC_WAIT_ALL();
  __syncthreads();

  int buf = 0;
  for (int k0 = 0; k0 < K; k0 += 32) {
    const bool has_next = (k0 + 32) < K;
    if (has_next) stage_load(buf ^ 1, k0 + 32);   // overlap DMA with compute

    const unsigned short* La = lds_a[buf];
    const unsigned short* Lb = lds_b[buf];
    Frag16 afr[4];
    for (int mt = 0; mt < 4; ++mt) {
      const int row = wm + mt * 16 + nlane;
      afr[mt].q[0] = *(const uint4*)&La[row * LDT + aoff];
      afr[mt].q[1] = *(const uint4*)&La[row * LDT + aoff + 16];
    }
    for (int nt = 0; nt < 4; ++nt) {
      Frag16 bfr;
      const int col = wn + nt * 16 + nlane;
      bfr.q[0] = *(const uint4*)&Lb[col * LDT + aoff];
      bfr.q[1] = *(const uint4*)&Lb[col * LDT + aoff + 16];
      for (int mt = 0; mt < 4; ++mt) {
        acc[mt][nt] = __builtin_amdgcn_wmma_f32_16x16x32_bf16(
            false, afr[mt].v, false, bfr.v, (short)0, acc[mt][nt],
            false, false);
      }
    }

    if (has_next) stage_commit(buf ^ 1);
    ASYNC_WAIT_ALL();        // next-stage DMA landed (async loads are in-order)
    __syncthreads();         // everyone done computing this stage
    buf ^= 1;
  }

  for (int mt = 0; mt < 4; ++mt)
    for (int nt = 0; nt < 4; ++nt)
      for (int r = 0; r < 8; ++r) {
        const int gm = m0 + wm + mt * 16 + mbase + r;
        const int gn = n0 + wn + nt * 16 + nlane;
        const float v = acc[mt][nt][r];
        if (OUT_BF16)
          ((unsigned short*)Cout)[(size_t)gm * N + gn] = f32_to_bf16(v);
        else
          ((float*)Cout)[(size_t)gm * N + gn] = v;
      }
}

// ---------------------------------------------------------------------------
// Flash attention per (b, h, 128-row Q block). Q fragments in registers,
// 64-key K/V blocks staged in LDS (V stored transposed for PV B-fragments),
// online softmax with half-wave shfl_xor row reductions, P round-trips a
// per-wave LDS scratch (LDS is in-order within a wave -> no barrier).
// ---------------------------------------------------------------------------
__global__ __launch_bounds__(256)
void mla_flash_attn(const unsigned short* __restrict__ Q,
                    const unsigned short* __restrict__ Kb,
                    const unsigned short* __restrict__ Vb,
                    unsigned short* __restrict__ Ctx) {
  constexpr int LKV = 136;  // 272B rows (16B multiple)
  constexpr int LVT = 72;   // 144B rows
  constexpr int LP  = 72;
  __shared__ unsigned short k_lds[64 * LKV];
  __shared__ unsigned short vt_lds[128 * LVT];
  __shared__ unsigned short p_lds[8 * 16 * LP];

  const int tid   = threadIdx.x;
  const int lane  = tid & 31;
  const int wave  = tid >> 5;
  const int nlane = lane & 15;
  const int hi    = (lane >= 16);
  const int mbase = hi ? 8 : 0;
  const int aoff  = hi ? 8 : 0;
  const int boff  = hi ? 16 : 0;

  const int s0   = blockIdx.x * 128;    // Q block
  const int h    = blockIdx.y;
  const int b    = blockIdx.z;
  const size_t rowbase = (size_t)b * SEQ * DIM;
  const int hcol = h * DHEAD;

  // Q fragments for this wave's 16 rows (4 K-steps of 16x32)
  Frag16 qf[4];
  {
    const int qrow = s0 + wave * 16 + nlane;
    const unsigned short* qp = &Q[rowbase + (size_t)qrow * DIM + hcol];
    for (int t = 0; t < 4; ++t) {
      qf[t].q[0] = *(const uint4*)&qp[t * 32 + aoff];
      qf[t].q[1] = *(const uint4*)&qp[t * 32 + aoff + 16];
    }
  }

  float mrow[8], lsum[8];
  v8f oacc[8];
  for (int r = 0; r < 8; ++r) { mrow[r] = -3.0e38f; lsum[r] = 0.f; }
  for (int nt = 0; nt < 8; ++nt) oacc[nt] = v8f_zero();

  const float scale = 0.08838834764831845f;  // 1/sqrt(128)

  const int krow   = tid >> 2;         // 0..63
  const int kchunk = (tid & 3) * 32;   // 0,32,64,96

  for (int j0 = 0; j0 < SEQ; j0 += 64) {
    const unsigned short* kp =
        &Kb[rowbase + (size_t)(j0 + krow) * DIM + hcol + kchunk];
    const unsigned short* vp =
        &Vb[rowbase + (size_t)(j0 + krow) * DIM + hcol + kchunk];
    uint4 kq[4], vq[4];
    for (int c = 0; c < 4; ++c) {
      kq[c] = *(const uint4*)&kp[c * 8];
      vq[c] = *(const uint4*)&vp[c * 8];
    }
    __syncthreads();
    for (int c = 0; c < 4; ++c)
      *(uint4*)&k_lds[krow * LKV + kchunk + c * 8] = kq[c];
    for (int c = 0; c < 4; ++c) {            // V transposed: vt[d][key]
      Pack8 pk; pk.q = vq[c];
      for (int e = 0; e < 8; ++e)
        vt_lds[(kchunk + c * 8 + e) * LVT + krow] = pk.s[e];
    }
    __syncthreads();

    // S = Q K^T (4 key n-tiles x 4 K-steps)
    v8f sfr[4];
    for (int nt = 0; nt < 4; ++nt) {
      v8f s = v8f_zero();
      const int kn = nt * 16 + nlane;
      for (int t = 0; t < 4; ++t) {
        Frag16 kf;
        kf.q[0] = *(const uint4*)&k_lds[kn * LKV + t * 32 + aoff];
        kf.q[1] = *(const uint4*)&k_lds[kn * LKV + t * 32 + aoff + 16];
        s = __builtin_amdgcn_wmma_f32_16x16x32_bf16(
            false, qf[t].v, false, kf.v, (short)0, s, false, false);
      }
      for (int r = 0; r < 8; ++r) s[r] *= scale;
      sfr[nt] = s;
    }

    // online softmax (rows replicated across each 16-lane half)
    float pr[4][8], corr[8];
    for (int r = 0; r < 8; ++r) {
      float mx = fmaxf(fmaxf(sfr[0][r], sfr[1][r]),
                       fmaxf(sfr[2][r], sfr[3][r]));
      mx = fmaxf(mx, __shfl_xor(mx, 1, 32));
      mx = fmaxf(mx, __shfl_xor(mx, 2, 32));
      mx = fmaxf(mx, __shfl_xor(mx, 4, 32));
      mx = fmaxf(mx, __shfl_xor(mx, 8, 32));
      const float mnew = fmaxf(mrow[r], mx);
      const float c = __expf(mrow[r] - mnew);
      float sum = 0.f;
      for (int nt = 0; nt < 4; ++nt) {
        const float p = __expf(sfr[nt][r] - mnew);
        pr[nt][r] = p;
        sum += p;
      }
      sum += __shfl_xor(sum, 1, 32);
      sum += __shfl_xor(sum, 2, 32);
      sum += __shfl_xor(sum, 4, 32);
      sum += __shfl_xor(sum, 8, 32);
      lsum[r] = lsum[r] * c + sum;
      mrow[r] = mnew;
      corr[r] = c;
    }
    for (int nt = 0; nt < 8; ++nt)
      for (int r = 0; r < 8; ++r) oacc[nt][r] *= corr[r];

    // P -> per-wave LDS scratch as bf16 (A-fragment source)
    unsigned short* pw = &p_lds[wave * 16 * LP];
    for (int nt = 0; nt < 4; ++nt)
      for (int r = 0; r < 8; ++r)
        pw[(mbase + r) * LP + nt * 16 + nlane] = f32_to_bf16(pr[nt][r]);

    // O += P V  (2 K-steps of 32 over the 64 keys, 8 output n-tiles)
    for (int kt = 0; kt < 2; ++kt) {
      Frag16 pf;
      pf.q[0] = *(const uint4*)&pw[nlane * LP + kt * 32 + aoff];
      pf.q[1] = *(const uint4*)&pw[nlane * LP + kt * 32 + aoff + 16];
      for (int nt = 0; nt < 8; ++nt) {
        Frag16 vf;
        const int vn = nt * 16 + nlane;
        vf.q[0] = *(const uint4*)&vt_lds[vn * LVT + kt * 32 + aoff];
        vf.q[1] = *(const uint4*)&vt_lds[vn * LVT + kt * 32 + aoff + 16];
        oacc[nt] = __builtin_amdgcn_wmma_f32_16x16x32_bf16(
            false, pf.v, false, vf.v, (short)0, oacc[nt], false, false);
      }
    }
  }

  // epilogue: O / l, store bf16 context [B,S,H*Dl]
  for (int r = 0; r < 8; ++r) lsum[r] = 1.0f / lsum[r];
  for (int nt = 0; nt < 8; ++nt)
    for (int r = 0; r < 8; ++r) {
      const int gm = s0 + wave * 16 + mbase + r;
      const int gn = hcol + nt * 16 + nlane;
      Ctx[rowbase + (size_t)gm * DIM + gn] = f32_to_bf16(oacc[nt][r] * lsum[r]);
    }
}

// ---------------------------------------------------------------------------
extern "C" void kernel_launch(void* const* d_in, const int* in_sizes, int n_in,
                              void* d_out, int out_size, void* d_ws,
                              size_t ws_size, hipStream_t stream) {
  const float* hs  = (const float*)d_in[0];  // [B,S,D]
  const float* Wq  = (const float*)d_in[1];  // [2048,2048]
  const float* Wkd = (const float*)d_in[2];
  const float* Wvd = (const float*)d_in[3];
  const float* Wku = (const float*)d_in[4];
  const float* Wo  = (const float*)d_in[5];

  char* ws = (char*)d_ws;
  const size_t actB = (size_t)MROWS * DIM * 2;  // 32MB bf16 activation
  const size_t wB   = (size_t)DIM * DIM * 2;    //  8MB bf16 weight
  unsigned short* hid_bf = (unsigned short*)(ws);
  unsigned short* wq_bf  = (unsigned short*)(ws + actB);
  unsigned short* wkd_bf = (unsigned short*)(ws + actB + 1 * wB);
  unsigned short* wvd_bf = (unsigned short*)(ws + actB + 2 * wB);
  unsigned short* wku_bf = (unsigned short*)(ws + actB + 3 * wB);
  unsigned short* wo_bf  = (unsigned short*)(ws + actB + 4 * wB);
  unsigned short* q_bf   = (unsigned short*)(ws + actB + 5 * wB);
  unsigned short* k_bf   = (unsigned short*)(ws + 2 * actB + 5 * wB);
  unsigned short* v_bf   = (unsigned short*)(ws + 3 * actB + 5 * wB);
  unsigned short* ctx_bf = (unsigned short*)(ws + 4 * actB + 5 * wB);
  unsigned short* up_bf  = (unsigned short*)(ws + 5 * actB + 5 * wB);

  const int nAct = MROWS * DIM;  // 16.7M
  const int nW   = DIM * DIM;    //  4.2M
  cvt_f32_bf16<<<4096, 256, 0, stream>>>(hs,  hid_bf, nAct);
  cvt_f32_bf16<<<2048, 256, 0, stream>>>(Wq,  wq_bf,  nW);
  cvt_f32_bf16<<<2048, 256, 0, stream>>>(Wkd, wkd_bf, nW);
  cvt_f32_bf16<<<2048, 256, 0, stream>>>(Wvd, wvd_bf, nW);
  cvt_f32_bf16<<<2048, 256, 0, stream>>>(Wku, wku_bf, nW);
  cvt_f32_bf16<<<2048, 256, 0, stream>>>(Wo,  wo_bf,  nW);

  const dim3 gg(DIM / 128, MROWS / 256);   // (16, 32)
  gemm_bf16_wmma<true><<<gg, 256, 0, stream>>>(hid_bf, wq_bf,  q_bf,
                                               MROWS, DIM, DIM);
  gemm_bf16_wmma<true><<<gg, 256, 0, stream>>>(hid_bf, wkd_bf, k_bf,
                                               MROWS, DIM, DIM);
  gemm_bf16_wmma<true><<<gg, 256, 0, stream>>>(hid_bf, wvd_bf, v_bf,
                                               MROWS, DIM, DIM);

  mla_flash_attn<<<dim3(SEQ / 128, NHEAD, NBAT), 256, 0, stream>>>(
      q_bf, k_bf, v_bf, ctx_bf);

  // faithful source bug: Wk_up applied to the attention output
  gemm_bf16_wmma<true><<<gg, 256, 0, stream>>>(ctx_bf, wku_bf, up_bf,
                                               MROWS, DIM, DIM);
  gemm_bf16_wmma<false><<<gg, 256, 0, stream>>>(up_bf, wo_bf, (float*)d_out,
                                                MROWS, DIM, DIM);
}
